// GraphDTA_GAT_82042465288997
// MI455X (gfx1250) — matrix operand order repaired and checked by reading, not compiled, not run on hardware
//
#include <hip/hip_runtime.h>
#include <hip/hip_bf16.h>

typedef __attribute__((ext_vector_type(16))) __bf16 bf16x16;
typedef __attribute__((ext_vector_type(8)))  float  f32x8;

#define NN    16384
#define BB    512
#define NPG_  32
#define PL_   1200
#define MAXD  64

__device__ __forceinline__ __bf16 f2bf(float f) { return (__bf16)f; }

__device__ __forceinline__ unsigned pkbf(float a, float b) {
    union { __bf16 h[2]; unsigned u; } x;
    x.h[0] = (__bf16)a; x.h[1] = (__bf16)b;
    return x.u;
}

// ---------------------------------------------------------------------------
// Generic WMMA GEMM: C[M,N] = act(A[M,K] @ B[K,N] + bias)
// block = 256 threads (8 waves), block tile 32(M) x 64(N), K step 32.
// Double-buffered LDS staging, one barrier per K-step. WMMA is issued BEFORE
// the next-tile staging so the ds-wait in front of it covers only the
// fragment loads; staging global loads then overlap WMMA execution.
// act: 0 = none, 1 = relu
// ---------------------------------------------------------------------------
__global__ __launch_bounds__(256)
void k_gemm(const float* __restrict__ A, const float* __restrict__ B,
            const float* __restrict__ bias, float* __restrict__ C,
            int M, int N, int K, int act)
{
    __shared__ __bf16 aLds[2][32][34];
    __shared__ __bf16 bLds[2][64][34];   // transposed: [n][k]
    const int t    = threadIdx.x;
    const int lane = t & 31, wv = t >> 5;
    const int hi   = lane >> 4, lo = lane & 15;
    const int mi   = wv >> 2, nj = wv & 3;
    const int m0   = blockIdx.y * 32, n0 = blockIdx.x * 64;
    const bool aFast = (m0 + 32 <= M) && ((K & 1) == 0);   // block-uniform
    const bool bFast = (n0 + 64 <= N) && ((N & 1) == 0);   // block-uniform

    auto stage = [&](int k0, int sb) {
        const bool kFull = (k0 + 32 <= K);                 // uniform
        {
            int r = t >> 3, kk = (t & 7) * 4;
            if (aFast && kFull) {
                const float* ap = A + (size_t)(m0 + r) * K + k0 + kk;
                float2 a0 = *(const float2*)ap;
                float2 a1 = *(const float2*)(ap + 2);
                *(unsigned*)&aLds[sb][r][kk]     = pkbf(a0.x, a0.y);
                *(unsigned*)&aLds[sb][r][kk + 2] = pkbf(a1.x, a1.y);
                if (k0 + 64 <= K) __builtin_prefetch(ap + 32, 0, 1);
            } else {
                int m  = m0 + r;
                int mc = (m < M) ? m : (M - 1);
                #pragma unroll
                for (int q = 0; q < 4; ++q) {
                    int k  = k0 + kk + q;
                    int kc = (k < K) ? k : (K - 1);
                    float v = A[(size_t)mc * K + kc];        // always in-bounds
                    aLds[sb][r][kk + q] = f2bf((m < M && k < K) ? v : 0.f);
                }
            }
        }
        {
            int kr = t >> 3, nc = (t & 7) * 8;
            if (bFast && kFull) {
                const float* bp = B + (size_t)(k0 + kr) * N + n0 + nc;
                #pragma unroll
                for (int q = 0; q < 4; ++q) {
                    float2 bv = *(const float2*)(bp + 2 * q);
                    bLds[sb][nc + 2 * q][kr]     = f2bf(bv.x);
                    bLds[sb][nc + 2 * q + 1][kr] = f2bf(bv.y);
                }
                if (k0 + 64 <= K) __builtin_prefetch(bp + (size_t)32 * N, 0, 1);
            } else {
                int k  = k0 + kr;
                int kc = (k < K) ? k : (K - 1);
                #pragma unroll
                for (int q = 0; q < 8; ++q) {
                    int n   = n0 + nc + q;
                    int ncl = (n < N) ? n : (N - 1);
                    float v = B[(size_t)kc * N + ncl];       // always in-bounds
                    bLds[sb][nc + q][kr] = f2bf((k < K && n < N) ? v : 0.f);
                }
            }
        }
    };

    f32x8 acc = {};
    stage(0, 0);
    int cur = 0;
    for (int k0 = 0; k0 < K; k0 += 32) {
        __syncthreads();
        bf16x16 af = {}, bfrag = {};
        int ar = mi * 16 + lo;
        #pragma unroll
        for (int v = 0; v < 8; ++v) {   // A 16x32: K = 16*hi + 2*v + p
            int kk = (hi << 4) + (v << 1);
            af[2 * v]     = aLds[cur][ar][kk];
            af[2 * v + 1] = aLds[cur][ar][kk + 1];
        }
        int bn = nj * 16 + lo;
        #pragma unroll
        for (int v = 0; v < 8; ++v) {   // B 32x16: K = 16*hi + 2*v + p
            int kk = (hi << 4) + (v << 1);
            bfrag[2 * v]     = bLds[cur][bn][kk];
            bfrag[2 * v + 1] = bLds[cur][bn][kk + 1];
        }
        acc = __builtin_amdgcn_wmma_f32_16x16x32_bf16(false, af, false, bfrag,
                                                      (short)0, acc, false, false);
        if (k0 + 32 < K) stage(k0 + 32, cur ^ 1);   // overlaps WMMA execution
        cur ^= 1;
    }
    int n = n0 + nj * 16 + lo;
    #pragma unroll
    for (int r = 0; r < 8; ++r) {       // C: M = 8*hi + r within tile
        int m = m0 + mi * 16 + hi * 8 + r;
        if (m < M && n < N) {
            float v = acc[r];
            if (bias) v += bias[n];
            if (act == 1) v = fmaxf(v, 0.f);
            C[(size_t)m * N + n] = v;
        }
    }
}

// ---------------------------------------------------------------------------
// Deterministic adjacency build: edges of graph g occupy [g*EPG, (g+1)*EPG)
// ---------------------------------------------------------------------------
__global__ void k_build_adj(const int* __restrict__ dst, int* __restrict__ eids,
                            int* __restrict__ deg, int E, int EPG)
{
    int n = blockIdx.x * blockDim.x + threadIdx.x;
    if (n >= NN) return;
    int g = n / NPG_;
    int e0 = g * EPG;
    int cnt = 0;
    int* el = eids + (size_t)n * MAXD;
    for (int j = 0; j < EPG; ++j) {
        int e = e0 + j;
        if (dst[e] == n && cnt < MAXD - 1) el[cnt++] = e;
    }
    el[cnt++] = E + n;   // self loop
    deg[n] = cnt;
}

// es[n,h] = <h[n,h,:], a_s[h,:]>, ed likewise
__global__ void k_gat_logits(const float* __restrict__ h, const float* __restrict__ as_,
                             const float* __restrict__ ad_, float* __restrict__ es,
                             float* __restrict__ ed, int H, int C)
{
    int i = blockIdx.x * blockDim.x + threadIdx.x;
    if (i >= NN * H) return;
    int n = i / H, hh = i % H;
    const float* hp  = h + (size_t)n * H * C + (size_t)hh * C;
    const float* sp  = as_ + (size_t)hh * C;
    const float* dp  = ad_ + (size_t)hh * C;
    float s = 0.f, d = 0.f;
    for (int c = 0; c < C; ++c) { s += hp[c] * sp[c]; d += hp[c] * dp[c]; }
    es[i] = s; ed[i] = d;
}

// ---------------------------------------------------------------------------
// Gather-style GAT softmax aggregation: one wave per destination node.
// alpha computed once per (edge, head) into LDS; feature loop is pure fmas.
// act: 0 = ELU, 1 = ReLU.  grid must be exactly NN/8 blocks of 256 threads.
// ---------------------------------------------------------------------------
__global__ __launch_bounds__(256)
void k_gat_agg(const float* __restrict__ h, const float* __restrict__ es,
               const float* __restrict__ ed, const int* __restrict__ eids,
               const int* __restrict__ deg, const int* __restrict__ esrc,
               const float* __restrict__ bias, float* __restrict__ out,
               int H, int C, int E, int act)
{
    __shared__ float alds[8][MAXD][10];
    __shared__ int   slds[8][MAXD];
    int lane = threadIdx.x & 31, wv = threadIdx.x >> 5;
    int n = blockIdx.x * 8 + wv;
    int dg = deg[n];
    const int* el = eids + (size_t)n * MAXD;
    for (int j = lane; j < dg; j += 32) {
        int e = el[j];
        slds[wv][j] = (e < E) ? esrc[e] : n;
    }
    __syncthreads();
    // per-head max / denom / alpha (lane l handles head l if l < H, else head 0)
    int hh0 = (lane < H) ? lane : 0;
    float edv = ed[(size_t)n * H + hh0];
    float mx = -1e30f;
    for (int j = 0; j < dg; ++j) {
        float lv = es[(size_t)slds[wv][j] * H + hh0] + edv;
        lv = (lv > 0.f) ? lv : 0.2f * lv;
        mx = fmaxf(mx, lv);
    }
    float den = 0.f;
    for (int j = 0; j < dg; ++j) {
        float lv = es[(size_t)slds[wv][j] * H + hh0] + edv;
        lv = (lv > 0.f) ? lv : 0.2f * lv;
        den += __expf(lv - mx);
    }
    float rden = 1.f / den;
    if (lane < H) {
        for (int j = 0; j < dg; ++j) {
            float lv = es[(size_t)slds[wv][j] * H + lane] + edv;
            lv = (lv > 0.f) ? lv : 0.2f * lv;
            alds[wv][j][lane] = __expf(lv - mx) * rden;
        }
    }
    __syncthreads();
    int HC = H * C;
    for (int fb = 0; fb < HC; fb += 32) {
        int f  = fb + lane;
        int ff = (f < HC) ? f : HC - 1;
        int hh = ff / C;
        float accv = 0.f;
        for (int j = 0; j < dg; ++j)
            accv += alds[wv][j][hh] * h[(size_t)slds[wv][j] * HC + ff];
        if (f < HC) {
            float v = accv + bias[f];
            v = (act == 1) ? fmaxf(v, 0.f) : ((v > 0.f) ? v : (__expf(v) - 1.f));
            out[(size_t)n * HC + f] = v;
        }
    }
}

// BatchNorm stats: one block per 64-column tile, coalesced row-major reads
__global__ __launch_bounds__(256)
void k_bn_stats(const float* __restrict__ x, float* __restrict__ mean,
                float* __restrict__ rsig, int C)
{
    __shared__ float ssum[4][64];
    __shared__ float ssq[4][64];
    int cl = threadIdx.x & 63;
    int rg = threadIdx.x >> 6;
    int c  = blockIdx.x * 64 + cl;
    float s = 0.f, q = 0.f;
    if (c < C) {
        for (int r = rg; r < NN; r += 4) {
            float v = x[(size_t)r * C + c];
            s += v; q += v * v;
        }
    }
    ssum[rg][cl] = s; ssq[rg][cl] = q;
    __syncthreads();
    if (rg == 0 && c < C) {
        float ts = ssum[0][cl] + ssum[1][cl] + ssum[2][cl] + ssum[3][cl];
        float tq = ssq[0][cl]  + ssq[1][cl]  + ssq[2][cl]  + ssq[3][cl];
        float mu  = ts / (float)NN;
        float var = tq / (float)NN - mu * mu;
        mean[c] = mu;
        rsig[c] = rsqrtf(var + 1e-5f);
    }
}

__global__ void k_bn_apply(float* __restrict__ x, const float* __restrict__ mean,
                           const float* __restrict__ rsig, const float* __restrict__ g,
                           const float* __restrict__ be, int C)
{
    size_t i = blockIdx.x * (size_t)blockDim.x + threadIdx.x;
    if (i >= (size_t)NN * C) return;
    int c = (int)(i % C);
    x[i] = (x[i] - mean[c]) * rsig[c] * g[c] + be[c];
}

// global max pool over NPG_ consecutive nodes, F = 128
__global__ void k_pool(const float* __restrict__ h, float* __restrict__ d)
{
    int b = blockIdx.x, f = threadIdx.x;
    const float* hp = h + (size_t)b * NPG_ * 128;
    float m = -1e30f;
    for (int j = 0; j < NPG_; ++j) m = fmaxf(m, hp[j * 128 + f]);
    d[b * 128 + f] = m;
}

// ---------------------------------------------------------------------------
// Protein conv1 as per-batch WMMA GEMM (M=32 oc, N=121 pos, K=9600) with
// on-the-fly im2col from the 26x128 embedding table in LDS. Double-buffered
// weight-tile staging issued after the WMMAs to overlap their execution.
// ---------------------------------------------------------------------------
__global__ __launch_bounds__(256)
void k_conv1(const float* __restrict__ emb, const int* __restrict__ target,
             const float* __restrict__ w, const float* __restrict__ bias,
             float* __restrict__ y)
{
    __shared__ __bf16 eLds[26][136];      // emb rows, zero-padded cols 128..135
    __shared__ __bf16 aLds[2][32][34];    // 32 oc x 32 kk, double-buffered
    __shared__ int    tk[2][4];
    int b = blockIdx.x, t = threadIdx.x;
    for (int idx = t * 2; idx < 26 * 128; idx += 512) {
        int row = idx >> 7, col = idx & 127;
        float2 ev = *(const float2*)(emb + row * 128 + col);
        *(unsigned*)&eLds[row][col] = pkbf(ev.x, ev.y);
    }
    for (int i = t; i < 26 * 8; i += 256) eLds[i >> 3][128 + (i & 7)] = f2bf(0.f);
    const int* tg = target + (size_t)b * PL_;

    auto stage = [&](int c0, int sb) {
        int o = t >> 3, q = t & 7;
        const float* wp = w + (size_t)o * PL_ * 8 + (size_t)c0 * 8 + q * 4;
        float4 w4 = *(const float4*)wp;          // 16B-aligned by layout
        *(unsigned*)&aLds[sb][o][q * 4]     = pkbf(w4.x, w4.y);
        *(unsigned*)&aLds[sb][o][q * 4 + 2] = pkbf(w4.z, w4.w);
        if (t < 4) {
            int v = tg[c0 + t];
            tk[sb][t] = (v < 0) ? 0 : (v > 25 ? 25 : v);
        }
    };

    int lane = t & 31, wv = t >> 5;
    int hi = lane >> 4, lo = lane & 15;
    int mi = wv >> 2, nj0 = wv & 3;    // wave owns n-tiles nj0 and nj0+4
    f32x8 acc0 = {}, acc1 = {};
    stage(0, 0);
    int cur = 0;
    for (int c0 = 0; c0 < PL_; c0 += 4) {
        __syncthreads();
        bf16x16 af = {}, b0 = {}, b1 = {};
        int ar = mi * 16 + lo;
        #pragma unroll
        for (int v = 0; v < 8; ++v) {
            int kk = (hi << 4) + (v << 1);
            af[2 * v]     = aLds[cur][ar][kk];
            af[2 * v + 1] = aLds[cur][ar][kk + 1];
        }
        #pragma unroll
        for (int v = 0; v < 8; ++v) {
            int kk = (hi << 4) + (v << 1);
            int cs = kk >> 3, ko = kk & 7;       // (c index, kernel offset)
            int row = tk[cur][cs];
            int n0 = nj0 * 16 + lo;
            b0[2 * v]     = eLds[row][n0 + ko];
            b0[2 * v + 1] = eLds[row][n0 + ko + 1];
            int n1 = n0 + 64;
            b1[2 * v]     = eLds[row][n1 + ko];
            b1[2 * v + 1] = eLds[row][n1 + ko + 1];
        }
        acc0 = __builtin_amdgcn_wmma_f32_16x16x32_bf16(false, af, false, b0,
                                                       (short)0, acc0, false, false);
        acc1 = __builtin_amdgcn_wmma_f32_16x16x32_bf16(false, af, false, b1,
                                                       (short)0, acc1, false, false);
        if (c0 + 4 < PL_) stage(c0 + 4, cur ^ 1);   // overlaps WMMA execution
        cur ^= 1;
    }
    #pragma unroll
    for (int r = 0; r < 8; ++r) {
        int o  = mi * 16 + hi * 8 + r;
        int l0 = nj0 * 16 + lo;
        float bb = bias[o];
        if (l0 < 121) y[((size_t)b * 32 + o) * 121 + l0] = fmaxf(acc0[r] + bb, 0.f);
        int l1 = l0 + 64;
        if (l1 < 121) y[((size_t)b * 32 + o) * 121 + l1] = fmaxf(acc1[r] + bb, 0.f);
    }
}

// ---------------------------------------------------------------------------
// Protein conv2 as per-batch WMMA GEMM (M=64 oc, N=114 pos, K=256), full
// weight tensor and the batch's y1 tile staged in LDS (bf16).
// ---------------------------------------------------------------------------
__global__ __launch_bounds__(256)
void k_conv2(const float* __restrict__ y1, const float* __restrict__ w,
             const float* __restrict__ bias, float* __restrict__ y2)
{
    __shared__ __bf16 wLds[64][260];   // [oc][ic*8 + k]
    __shared__ __bf16 xLds[32][136];   // [ic][l], zero-padded l >= 121
    int b = blockIdx.x, t = threadIdx.x;
    for (int base = t * 4; base < 64 * 256; base += 1024) {
        float4 w4 = *(const float4*)(w + base);      // 16B-aligned
        int r = base >> 8, c = base & 255;
        *(unsigned*)&wLds[r][c]     = pkbf(w4.x, w4.y);
        *(unsigned*)&wLds[r][c + 2] = pkbf(w4.z, w4.w);
    }
    for (int i = t; i < 32 * 136; i += 256) {
        int r = i / 136, c = i % 136;
        int cc = (c < 121) ? c : 120;
        float v = y1[((size_t)b * 32 + r) * 121 + cc];   // always in-bounds
        xLds[r][c] = f2bf((c < 121) ? v : 0.f);
    }
    __syncthreads();
    int lane = t & 31, wv = t >> 5;
    int hi = lane >> 4, lo = lane & 15;
    int mi = wv >> 1, njb = wv & 1;    // wave owns n-tiles njb, +2, +4, +6
    f32x8 acc[4] = {};
    for (int s = 0; s < 8; ++s) {
        bf16x16 af = {};
        int ar = mi * 16 + lo;
        #pragma unroll
        for (int v = 0; v < 8; ++v) {
            int kk = s * 32 + (hi << 4) + (v << 1);
            af[2 * v]     = wLds[ar][kk];
            af[2 * v + 1] = wLds[ar][kk + 1];
        }
        #pragma unroll
        for (int u = 0; u < 4; ++u) {
            int nj = njb + 2 * u;
            int l = nj * 16 + lo;
            bf16x16 bfrag = {};
            #pragma unroll
            for (int v = 0; v < 8; ++v) {
                int kk = s * 32 + (hi << 4) + (v << 1);
                int ic = kk >> 3, ko = kk & 7;
                bfrag[2 * v]     = xLds[ic][l + ko];
                bfrag[2 * v + 1] = xLds[ic][l + ko + 1];
            }
            acc[u] = __builtin_amdgcn_wmma_f32_16x16x32_bf16(false, af, false, bfrag,
                                                             (short)0, acc[u], false, false);
        }
    }
    #pragma unroll
    for (int u = 0; u < 4; ++u) {
        int nj = njb + 2 * u;
        int l = nj * 16 + lo;
        #pragma unroll
        for (int r = 0; r < 8; ++r) {
            int oc = mi * 16 + hi * 8 + r;
            if (l < 114)
                y2[(size_t)b * 7296 + oc * 114 + l] = fmaxf(acc[u][r] + bias[oc], 0.f);
        }
    }
}

__global__ void k_concat(const float* __restrict__ a, const float* __restrict__ b,
                         float* __restrict__ c)
{
    int i = blockIdx.x * blockDim.x + threadIdx.x;
    if (i >= BB * 256) return;
    int bb = i >> 8, f = i & 255;
    c[i] = (f < 128) ? a[bb * 128 + f] : b[bb * 128 + (f - 128)];
}

// ---------------------------------------------------------------------------
extern "C" void kernel_launch(void* const* d_in, const int* in_sizes, int n_in,
                              void* d_out, int out_size, void* d_ws, size_t ws_size,
                              hipStream_t stream)
{
    (void)n_in; (void)out_size; (void)ws_size;
    const float* x      = (const float*)d_in[0];
    const int*   eidx   = (const int*)d_in[1];
    const int*   target = (const int*)d_in[3];
    const float* W1  = (const float*)d_in[4];
    const float* as1 = (const float*)d_in[5];
    const float* ad1 = (const float*)d_in[6];
    const float* b1  = (const float*)d_in[7];
    const float* g1  = (const float*)d_in[8];
    const float* be1 = (const float*)d_in[9];
    const float* W2  = (const float*)d_in[10];
    const float* as2 = (const float*)d_in[11];
    const float* ad2 = (const float*)d_in[12];
    const float* b2  = (const float*)d_in[13];
    const float* g2  = (const float*)d_in[14];
    const float* be2 = (const float*)d_in[15];
    const float* W3  = (const float*)d_in[16];
    const float* as3 = (const float*)d_in[17];
    const float* ad3 = (const float*)d_in[18];
    const float* b3  = (const float*)d_in[19];
    const float* g3  = (const float*)d_in[20];
    const float* be3 = (const float*)d_in[21];
    const float* fw1 = (const float*)d_in[22];
    const float* fb1 = (const float*)d_in[23];
    const float* fw2 = (const float*)d_in[24];
    const float* fb2 = (const float*)d_in[25];
    const float* emb = (const float*)d_in[26];
    const float* cw1 = (const float*)d_in[27];
    const float* cb1 = (const float*)d_in[28];
    const float* cw2 = (const float*)d_in[29];
    const float* cb2 = (const float*)d_in[30];
    const float* pw  = (const float*)d_in[31];
    const float* pb  = (const float*)d_in[32];
    const float* ow1 = (const float*)d_in[33];
    const float* ob1 = (const float*)d_in[34];
    const float* ow2 = (const float*)d_in[35];
    const float* ob2 = (const float*)d_in[36];
    const float* ow3 = (const float*)d_in[37];
    const float* ob3 = (const float*)d_in[38];

    const int E    = in_sizes[1] / 2;      // 131072
    const int EPG  = E / BB;               // 256 edges per graph
    const int* esrc = eidx;
    const int* edst = eidx + E;

    // ---- workspace carve-out ----
    char* base = (char*)d_ws;
    size_t off = 0;
    auto alloc = [&](size_t bytes) -> void* {
        void* p = base + off;
        off += (bytes + 255) & ~(size_t)255;
        return p;
    };
    float* hpre  = (float*)alloc(sizeof(float) * (size_t)NN * 780);
    float* hact  = (float*)alloc(sizeof(float) * (size_t)NN * 780);
    float* es    = (float*)alloc(sizeof(float) * (size_t)NN * 10);
    float* ed    = (float*)alloc(sizeof(float) * (size_t)NN * 10);
    float* mean  = (float*)alloc(sizeof(float) * 1024);
    float* rsig  = (float*)alloc(sizeof(float) * 1024);
    int*   eids  = (int*)alloc(sizeof(int) * (size_t)NN * MAXD);
    int*   deg   = (int*)alloc(sizeof(int) * NN);
    float* pooled= (float*)alloc(sizeof(float) * BB * 128);
    float* dt1   = (float*)alloc(sizeof(float) * BB * 256);
    float* dfeat = (float*)alloc(sizeof(float) * BB * 128);
    float* tfeat = (float*)alloc(sizeof(float) * BB * 128);
    float* cvec  = (float*)alloc(sizeof(float) * BB * 256);
    float* o1    = (float*)alloc(sizeof(float) * BB * 1024);
    float* o2    = (float*)alloc(sizeof(float) * BB * 256);
    // protein intermediates reuse hpre (free once GAT stack is done)
    float* y1 = hpre;               // 512*32*121 = 1.98M floats
    float* y2 = hpre + 4000000;     // 512*7296   = 3.74M floats

    auto gemm = [&](const float* A, const float* Bm, const float* bias, float* C,
                    int M, int N, int K, int act) {
        dim3 g((N + 63) / 64, (M + 31) / 32);
        k_gemm<<<g, 256, 0, stream>>>(A, Bm, bias, C, M, N, K, act);
    };

    // adjacency (deterministic, reused by all 3 GAT layers)
    k_build_adj<<<NN / 256, 256, 0, stream>>>(edst, eids, deg, E, EPG);

    // ---- GAT layer 1: 18 -> 78, 1 head, ELU, BN ----
    gemm(x, W1, nullptr, hpre, NN, 78, 18, 0);
    k_gat_logits<<<(NN * 1 + 255) / 256, 256, 0, stream>>>(hpre, as1, ad1, es, ed, 1, 78);
    k_gat_agg<<<NN / 8, 256, 0, stream>>>(hpre, es, ed, eids, deg, esrc, b1, hact, 1, 78, E, 0);
    k_bn_stats<<<(78 + 63) / 64, 256, 0, stream>>>(hact, mean, rsig, 78);
    k_bn_apply<<<((size_t)NN * 78 + 255) / 256, 256, 0, stream>>>(hact, mean, rsig, g1, be1, 78);

    // ---- GAT layer 2: 78 -> 780, 10 heads, ELU, BN ----
    gemm(hact, W2, nullptr, hpre, NN, 780, 78, 0);
    k_gat_logits<<<(NN * 10 + 255) / 256, 256, 0, stream>>>(hpre, as2, ad2, es, ed, 10, 78);
    k_gat_agg<<<NN / 8, 256, 0, stream>>>(hpre, es, ed, eids, deg, esrc, b2, hact, 10, 78, E, 0);
    k_bn_stats<<<(780 + 63) / 64, 256, 0, stream>>>(hact, mean, rsig, 780);
    k_bn_apply<<<((size_t)NN * 780 + 255) / 256, 256, 0, stream>>>(hact, mean, rsig, g2, be2, 780);

    // ---- GAT layer 3: 780 -> 128, 1 head, ReLU, BN ----
    gemm(hact, W3, nullptr, hpre, NN, 128, 780, 0);
    k_gat_logits<<<(NN * 1 + 255) / 256, 256, 0, stream>>>(hpre, as3, ad3, es, ed, 1, 128);
    k_gat_agg<<<NN / 8, 256, 0, stream>>>(hpre, es, ed, eids, deg, esrc, b3, hact, 1, 128, E, 1);
    k_bn_stats<<<(128 + 63) / 64, 256, 0, stream>>>(hact, mean, rsig, 128);
    k_bn_apply<<<((size_t)NN * 128 + 255) / 256, 256, 0, stream>>>(hact, mean, rsig, g3, be3, 128);

    // ---- pool + drug MLP ----
    k_pool<<<BB, 128, 0, stream>>>(hact, pooled);
    gemm(pooled, fw1, fb1, dt1, BB, 256, 128, 1);
    gemm(dt1, fw2, fb2, dfeat, BB, 128, 256, 0);

    // ---- protein branch ----
    k_conv1<<<BB, 256, 0, stream>>>(emb, target, cw1, cb1, y1);
    k_conv2<<<BB, 256, 0, stream>>>(y1, cw2, cb2, y2);
    gemm(y2, pw, pb, tfeat, BB, 128, 7296, 0);

    // ---- concat + output MLP ----
    k_concat<<<(BB * 256 + 255) / 256, 256, 0, stream>>>(dfeat, tfeat, cvec);
    gemm(cvec, ow1, ob1, o1, BB, 1024, 256, 1);
    gemm(o1, ow2, ob2, o2, BB, 256, 1024, 1);
    gemm(o2, ow3, ob3, (float*)d_out, BB, 1, 256, 0);
}
